// MultiHeadSelfAttention_40716289966214
// MI455X (gfx1250) — matrix operand order repaired and checked by reading, not compiled, not run on hardware
//
#include <hip/hip_runtime.h>
#include <hip/hip_bf16.h>

typedef __bf16 bf16;
typedef __attribute__((ext_vector_type(16))) __bf16 v16bf;
typedef __attribute__((ext_vector_type(8)))  __bf16 v8bf;
typedef __attribute__((ext_vector_type(8)))  float  v8f;
typedef __attribute__((ext_vector_type(4)))  float  v4f;

#define EMB   1024
#define TSEQ  2048
#define NBAT  4
#define HEADS 16
#define HD    64
#define MTOT  (NBAT * TSEQ)   /* 8192 */
#define KT2   (EMB / 64)      /* 16 staged K-tiles of 64 */

static __device__ __forceinline__ v8f vzero8() {
  v8f v;
#pragma unroll
  for (int i = 0; i < 8; ++i) v[i] = 0.0f;
  return v;
}

// Build a 16-half A/B fragment from two 16-byte chunks.
static __device__ __forceinline__ v16bf ld_frag(const bf16* p0, const bf16* p1) {
  v16bf r;
  *reinterpret_cast<v8bf*>(&r)       = *reinterpret_cast<const v8bf*>(p0);
  *(reinterpret_cast<v8bf*>(&r) + 1) = *reinterpret_cast<const v8bf*>(p1);
  return r;
}

static __device__ __forceinline__ v8f wmma_bf16(v16bf a, v16bf b, v8f c) {
  // (neg_a, A, neg_b, B, c_mod, C, reuse_a, reuse_b)
  return __builtin_amdgcn_wmma_f32_16x16x32_bf16(false, a, false, b, (short)0, c,
                                                 false, false);
}

// Wave-relative LDS byte offset of a flat shared pointer (aperture translation
// keeps addr[31:0] as the LDS address).
static __device__ __forceinline__ unsigned lds_off(const void* p) {
  return (unsigned)(unsigned long long)p;
}

static __device__ __forceinline__ void async_ld_b128(unsigned lds_addr,
                                                     unsigned long long gaddr) {
  asm volatile("global_load_async_to_lds_b128 %0, %1, off"
               :: "v"(lds_addr), "v"(gaddr) : "memory");
}

// ---------------------------------------------------------------------------
// f32 -> bf16 conversion (8 elements / thread, b128 in / b128 out)
// ---------------------------------------------------------------------------
__global__ __launch_bounds__(256) void cvt_f32_to_bf16(const float* __restrict__ src,
                                                       bf16* __restrict__ dst,
                                                       int n8) {
  int i = blockIdx.x * 256 + threadIdx.x;
  if (i >= n8) return;
  const v4f* pv = reinterpret_cast<const v4f*>(src + (size_t)i * 8);
  v4f lo = pv[0], hi = pv[1];
  v8bf o;
#pragma unroll
  for (int j = 0; j < 4; ++j) { o[j] = (bf16)lo[j]; o[4 + j] = (bf16)hi[j]; }
  *reinterpret_cast<v8bf*>(dst + (size_t)i * 8) = o;
}

// ---------------------------------------------------------------------------
// Shared GEMM core: C(64x128 per block) = A[M,K] @ W[N,K]^T, bf16 in, f32 acc.
// K staged in 64-wide tiles via async copy into LDS, double buffered.
// 4 waves; each wave computes a 32x64 C-area (16 WMMAs per staged tile).
// ---------------------------------------------------------------------------
struct GemmSmem {
  bf16 A[2][64 * 64];     // [buf][row][k]   8 KiB per buf
  bf16 B[2][128 * 64];    // [buf][row][k]  16 KiB per buf
};

static __device__ __forceinline__ void gemm_core(const bf16* __restrict__ Xg,
                                                 const bf16* __restrict__ Wg,
                                                 int m0blk, int n0blk,
                                                 GemmSmem& sm, v8f c[2][4]) {
  const int tid  = threadIdx.x;
  const int lane = tid & 31;
  const int wid  = tid >> 5;
  const int wm = wid & 1, wn = wid >> 1;
  const int lm = lane & 15, half = lane >> 4;

  // --- async-copy lane mapping (16B chunks; 8 chunks per 64-elem row) ------
  unsigned a_lds[4]; unsigned long long a_g[4];
#pragma unroll
  for (int i = 0; i < 4; ++i) {
    const int id = wid * 128 + i * 32 + lane;     // 512 chunks, 64x64 A tile
    const int r = id >> 3, ch = id & 7;
    a_lds[i] = lds_off(&sm.A[0][r * 64 + ch * 8]);
    a_g[i] = (unsigned long long)(const void*)(Xg + (size_t)(m0blk + r) * EMB + ch * 8);
  }
  unsigned b_lds[8]; unsigned long long b_g[8];
#pragma unroll
  for (int i = 0; i < 8; ++i) {
    const int id = wid * 256 + i * 32 + lane;     // 1024 chunks, 128x64 B tile
    const int r = id >> 3, ch = id & 7;
    b_lds[i] = lds_off(&sm.B[0][r * 64 + ch * 8]);
    b_g[i] = (unsigned long long)(const void*)(Wg + (size_t)(n0blk + r) * EMB + ch * 8);
  }
  int adelta = 64 * 64 * 2;     // bytes between A buffers (alternating +/-)
  int bdelta = 128 * 64 * 2;    // bytes between B buffers

  auto issue = [&](int kt) {
    const unsigned long long ko = (unsigned long long)kt * 128;  // 64 bf16 = 128 B
#pragma unroll
    for (int i = 0; i < 4; ++i) async_ld_b128(a_lds[i], a_g[i] + ko);
#pragma unroll
    for (int i = 0; i < 8; ++i) async_ld_b128(b_lds[i], b_g[i] + ko);
#pragma unroll
    for (int i = 0; i < 4; ++i) a_lds[i] += adelta;
#pragma unroll
    for (int i = 0; i < 8; ++i) b_lds[i] += bdelta;
    adelta = -adelta;
    bdelta = -bdelta;
  };

#pragma unroll
  for (int i = 0; i < 2; ++i)
#pragma unroll
    for (int j = 0; j < 4; ++j) c[i][j] = vzero8();

  const bf16* Acur = &sm.A[0][0];
  const bf16* Anxt = &sm.A[1][0];
  const bf16* Bcur = &sm.B[0][0];
  const bf16* Bnxt = &sm.B[1][0];

  issue(0);
#pragma unroll 1
  for (int kt = 0; kt < KT2; ++kt) {
    if (kt + 1 < KT2) {
      issue(kt + 1);
      // previous batch (12 ops) done when <= 12 newest remain outstanding
      asm volatile("s_wait_asynccnt 0xC" ::: "memory");
    } else {
      asm volatile("s_wait_asynccnt 0x0" ::: "memory");
    }
    __syncthreads();  // staged tile fully visible to all 4 waves

    const bf16* a_base = Acur + (wm * 32 + lm) * 64 + half * 8;
    const bf16* b_base = Bcur + (wn * 64 + lm) * 64 + half * 8;
#pragma unroll
    for (int ks = 0; ks < 2; ++ks) {
      const bf16* ap = a_base + ks * 32;
      v16bf a0 = ld_frag(ap, ap + 16);
      v16bf a1 = ld_frag(ap + 16 * 64, ap + 16 * 64 + 16);
#pragma unroll
      for (int j = 0; j < 4; ++j) {
        const bf16* bp = b_base + j * 16 * 64 + ks * 32;
        v16bf b = ld_frag(bp, bp + 16);
        c[0][j] = wmma_bf16(a0, b, c[0][j]);
        c[1][j] = wmma_bf16(a1, b, c[1][j]);
      }
    }
    __syncthreads();  // all waves done reading before buffer reuse

    const bf16* t0 = Acur; Acur = Anxt; Anxt = t0;
    const bf16* t1 = Bcur; Bcur = Bnxt; Bnxt = t1;
  }
}

// ---------------------------------------------------------------------------
// Fused QKV projection: y = x @ W^T + b.  blockIdx.z selects Q / K / V.
// Q,K stored [B,H,T,hd]; V stored transposed [B,H,hd,T].
// ---------------------------------------------------------------------------
__global__ __launch_bounds__(128) void qkv_proj_kernel(
    const bf16* __restrict__ X,
    const bf16* __restrict__ WQ, const bf16* __restrict__ WK, const bf16* __restrict__ WV,
    const float* __restrict__ bq, const float* __restrict__ bk, const float* __restrict__ bv,
    bf16* __restrict__ Qo, bf16* __restrict__ Ko, bf16* __restrict__ Vt) {
  const int z = blockIdx.z;
  const bf16*  W    = (z == 0) ? WQ : ((z == 1) ? WK : WV);
  const float* bias = (z == 0) ? bq : ((z == 1) ? bk : bv);

  __shared__ __align__(16) GemmSmem sm;

  const int lane = threadIdx.x & 31;
  const int wid  = threadIdx.x >> 5;
  const int wm = wid & 1, wn = wid >> 1;
  const int lm = lane & 15, half = lane >> 4;

  const int m0 = blockIdx.x * 64 + wm * 32;
  const int n0 = blockIdx.y * 128 + wn * 64;

  v8f c[2][4];
  gemm_core(X, W, blockIdx.x * 64, blockIdx.y * 128, sm, c);

  bf16* OQK = (z == 0) ? Qo : Ko;
#pragma unroll
  for (int i = 0; i < 2; ++i) {
    const int mrow = m0 + i * 16 + half * 8;   // first of 8 consecutive rows
    const int bb = mrow >> 11;                 // / 2048
    const int tt = mrow & (TSEQ - 1);
#pragma unroll
    for (int j = 0; j < 4; ++j) {
      const int n = n0 + j * 16 + lm;
      const float bv_ = bias[n];
      const int hh = n >> 6, dd = n & 63;
      if (z < 2) {
        bf16* p = OQK + ((((size_t)bb * HEADS + hh) * TSEQ + tt) * HD + dd);
#pragma unroll
        for (int r = 0; r < 8; ++r) p[(size_t)r * HD] = (bf16)(c[i][j][r] + bv_);
      } else {
        bf16* p = Vt + ((((size_t)bb * HEADS + hh) * HD + dd) * TSEQ + tt);
        v8bf pk;
#pragma unroll
        for (int r = 0; r < 8; ++r) pk[r] = (bf16)(c[i][j][r] + bv_);
        *reinterpret_cast<v8bf*>(p) = pk;      // 8 consecutive t -> one b128 store
      }
    }
  }
}

// ---------------------------------------------------------------------------
// Flash attention: one wave per 16-query tile, 32 keys per step.
// ctx written bf16 [B,T,E].
// ---------------------------------------------------------------------------
__global__ __launch_bounds__(128) void attn_kernel(const bf16* __restrict__ Q,
                                                   const bf16* __restrict__ Kd,
                                                   const bf16* __restrict__ Vt,
                                                   bf16* __restrict__ ctx) {
  const int lane = threadIdx.x & 31;
  const int wid  = threadIdx.x >> 5;
  const int lm = lane & 15, half = lane >> 4;

  const int bh = blockIdx.y;                       // 0..63 = b*16+h
  const int q0 = (blockIdx.x * 4 + wid) * 16;

  const bf16* Qh = Q  + (size_t)bh * TSEQ * HD;
  const bf16* Kh = Kd + (size_t)bh * TSEQ * HD;
  const bf16* Vh = Vt + (size_t)bh * HD * TSEQ;
  const int b = bh >> 4, h = bh & 15;
  bf16* ctxp = ctx + (size_t)b * TSEQ * EMB + h * HD;

  __shared__ __align__(16) bf16 plds[4][16 * 32];  // per-wave P transpose tile
  bf16* my = &plds[wid][0];

  const bf16* qp = Qh + (size_t)(q0 + lm) * HD + half * 8;
  v16bf qa0 = ld_frag(qp, qp + 16);        // d = 0..31
  v16bf qa1 = ld_frag(qp + 32, qp + 48);   // d = 32..63

  float mr[8], lr[8];
  v8f o[4];
#pragma unroll
  for (int r = 0; r < 8; ++r) { mr[r] = -__builtin_inff(); lr[r] = 0.0f; }
#pragma unroll
  for (int j = 0; j < 4; ++j) o[j] = vzero8();

  const float LOG2E = 1.4426950408889634f;

  for (int j0 = 0; j0 < TSEQ; j0 += 32) {
    // S = (Q K^T) / 8 for keys [j0, j0+32)
    const bf16* kp0 = Kh + (size_t)(j0 + lm) * HD + half * 8;
    const bf16* kp1 = kp0 + (size_t)16 * HD;
    v8f s0 = vzero8(), s1 = vzero8();
    {
      v16bf b00 = ld_frag(kp0, kp0 + 16);       // keys j0..+15, d 0..31
      v16bf b01 = ld_frag(kp0 + 32, kp0 + 48);  // d 32..63
      s0 = wmma_bf16(qa0, b00, s0);
      s0 = wmma_bf16(qa1, b01, s0);
      v16bf b10 = ld_frag(kp1, kp1 + 16);
      v16bf b11 = ld_frag(kp1 + 32, kp1 + 48);
      s1 = wmma_bf16(qa0, b10, s1);
      s1 = wmma_bf16(qa1, b11, s1);
    }

    // online softmax; row r of this lane is (r + 8*half)
#pragma unroll
    for (int r = 0; r < 8; ++r) {
      float sa = s0[r] * 0.125f;
      float sb = s1[r] * 0.125f;
      float v = fmaxf(sa, sb);
      v = fmaxf(v, __shfl_xor(v, 1, 32));
      v = fmaxf(v, __shfl_xor(v, 2, 32));
      v = fmaxf(v, __shfl_xor(v, 4, 32));
      v = fmaxf(v, __shfl_xor(v, 8, 32));
      float mn = fmaxf(mr[r], v);
      float alpha = exp2f((mr[r] - mn) * LOG2E);
      float p0 = exp2f((sa - mn) * LOG2E);
      float p1 = exp2f((sb - mn) * LOG2E);
      float rs = p0 + p1;
      rs += __shfl_xor(rs, 1, 32);
      rs += __shfl_xor(rs, 2, 32);
      rs += __shfl_xor(rs, 4, 32);
      rs += __shfl_xor(rs, 8, 32);
      lr[r] = lr[r] * alpha + rs;
      mr[r] = mn;
      const int row = r + half * 8;
      my[row * 32 + lm]      = (bf16)p0;   // C-layout -> LDS [16][32]
      my[row * 32 + 16 + lm] = (bf16)p1;
#pragma unroll
      for (int j = 0; j < 4; ++j) o[j][r] *= alpha;
    }

    // P as A-fragment (16x32) from LDS (same-wave DS ops are in-order)
    const bf16* pp = my + lm * 32 + half * 8;
    v16bf pa = ld_frag(pp, pp + 16);

    // O += P @ V  (Vt is [hd][T] so B-fragments are contiguous loads)
#pragma unroll
    for (int jd = 0; jd < 4; ++jd) {
      const bf16* vp = Vh + (size_t)(jd * 16 + lm) * TSEQ + j0 + half * 8;
      v16bf vb = ld_frag(vp, vp + 16);
      o[jd] = wmma_bf16(pa, vb, o[jd]);
    }
  }

  float inv[8];
#pragma unroll
  for (int r = 0; r < 8; ++r) inv[r] = 1.0f / lr[r];
#pragma unroll
  for (int jd = 0; jd < 4; ++jd) {
#pragma unroll
    for (int r = 0; r < 8; ++r) {
      const int t = q0 + r + half * 8;
      ctxp[(size_t)t * EMB + jd * 16 + lm] = (bf16)(o[jd][r] * inv[r]);
    }
  }
}

// ---------------------------------------------------------------------------
// Output projection: out = ctx @ Wo^T + bo  (f32 output, row-major [8192,1024])
// ---------------------------------------------------------------------------
__global__ __launch_bounds__(128) void out_proj_kernel(const bf16* __restrict__ CTX,
                                                       const bf16* __restrict__ WO,
                                                       const float* __restrict__ bo,
                                                       float* __restrict__ out) {
  __shared__ __align__(16) GemmSmem sm;

  const int lane = threadIdx.x & 31;
  const int wid  = threadIdx.x >> 5;
  const int wm = wid & 1, wn = wid >> 1;
  const int lm = lane & 15, half = lane >> 4;

  const int m0 = blockIdx.x * 64 + wm * 32;
  const int n0 = blockIdx.y * 128 + wn * 64;

  v8f c[2][4];
  gemm_core(CTX, WO, blockIdx.x * 64, blockIdx.y * 128, sm, c);

#pragma unroll
  for (int i = 0; i < 2; ++i) {
    const int mrow = m0 + i * 16 + half * 8;
#pragma unroll
    for (int j = 0; j < 4; ++j) {
      const int n = n0 + j * 16 + lm;
      const float bv_ = bo[n];
      float* op = out + (size_t)mrow * EMB + n;
#pragma unroll
      for (int r = 0; r < 8; ++r) op[(size_t)r * EMB] = c[i][j][r] + bv_;
    }
  }
}

// ---------------------------------------------------------------------------
extern "C" void kernel_launch(void* const* d_in, const int* in_sizes, int n_in,
                              void* d_out, int out_size, void* d_ws, size_t ws_size,
                              hipStream_t stream) {
  (void)in_sizes; (void)n_in; (void)out_size;

  const float* x  = (const float*)d_in[0];
  const float* Wq = (const float*)d_in[1];
  const float* bq = (const float*)d_in[2];
  const float* Wk = (const float*)d_in[3];
  const float* bk = (const float*)d_in[4];
  const float* Wv = (const float*)d_in[5];
  const float* bv = (const float*)d_in[6];
  const float* Wo = (const float*)d_in[7];
  const float* bo = (const float*)d_in[8];
  float* out = (float*)d_out;

  // Workspace layout (bytes)
  const size_t SZ_X  = (size_t)MTOT * EMB * 2;        // 16 MiB
  const size_t SZ_W  = (size_t)EMB * EMB * 2;         //  2 MiB each
  const size_t SZ_QK = (size_t)MTOT * EMB * 2;        // 16 MiB each
  const size_t OFF_XB  = 0;
  const size_t OFF_WQ  = OFF_XB + SZ_X;
  const size_t OFF_WK  = OFF_WQ + SZ_W;
  const size_t OFF_WV  = OFF_WK + SZ_W;
  const size_t OFF_WO  = OFF_WV + SZ_W;
  const size_t OFF_Q   = OFF_WO + SZ_W;
  const size_t OFF_K   = OFF_Q + SZ_QK;
  const size_t OFF_VT  = OFF_K + SZ_QK;
  const size_t OFF_CTX = OFF_VT + SZ_QK;
  const size_t TOTAL   = OFF_CTX + SZ_QK;             // 88 MiB
  if (ws_size < TOTAL) return;

  char* w = (char*)d_ws;
  bf16* xb   = (bf16*)(w + OFF_XB);
  bf16* wqb  = (bf16*)(w + OFF_WQ);
  bf16* wkb  = (bf16*)(w + OFF_WK);
  bf16* wvb  = (bf16*)(w + OFF_WV);
  bf16* wob  = (bf16*)(w + OFF_WO);
  bf16* Qb   = (bf16*)(w + OFF_Q);
  bf16* Kb   = (bf16*)(w + OFF_K);
  bf16* Vtb  = (bf16*)(w + OFF_VT);
  bf16* ctxb = (bf16*)(w + OFF_CTX);

  // 1) f32 -> bf16 conversions
  cvt_f32_to_bf16<<<(MTOT * EMB / 8) / 256, 256, 0, stream>>>(x, xb, MTOT * EMB / 8);
  cvt_f32_to_bf16<<<(EMB * EMB / 8) / 256, 256, 0, stream>>>(Wq, wqb, EMB * EMB / 8);
  cvt_f32_to_bf16<<<(EMB * EMB / 8) / 256, 256, 0, stream>>>(Wk, wkb, EMB * EMB / 8);
  cvt_f32_to_bf16<<<(EMB * EMB / 8) / 256, 256, 0, stream>>>(Wv, wvb, EMB * EMB / 8);
  cvt_f32_to_bf16<<<(EMB * EMB / 8) / 256, 256, 0, stream>>>(Wo, wob, EMB * EMB / 8);

  // 2) fused QKV projection (async-LDS staged, double buffered, BK=64)
  qkv_proj_kernel<<<dim3(MTOT / 64, EMB / 128, 3), 128, 0, stream>>>(
      xb, wqb, wkb, wvb, bq, bk, bv, Qb, Kb, Vtb);

  // 3) flash attention
  attn_kernel<<<dim3(TSEQ / 16 / 4, NBAT * HEADS), 128, 0, stream>>>(Qb, Kb, Vtb, ctxb);

  // 4) output projection
  out_proj_kernel<<<dim3(MTOT / 64, EMB / 128), 128, 0, stream>>>(ctxb, wob, bo, out);
}
